// MemoryBank_69999376990806
// MI455X (gfx1250) — matrix operand order repaired and checked by reading, not compiled, not run on hardware
//
#include <hip/hip_runtime.h>
#include <math.h>

// ---- CDNA5 WMMA vector types (wave32) ----
typedef __attribute__((ext_vector_type(16))) __bf16 v16bf;
typedef __attribute__((ext_vector_type(8)))  __bf16 v8bf;
typedef __attribute__((ext_vector_type(4)))  __bf16 v4bf;
typedef __attribute__((ext_vector_type(8)))  float  v8f;

#define B_N   1024
#define D_K   128
#define NCLS  50
#define KCAP  4096
#define NCOL  (NCLS * KCAP)            // 204800 bank rows
#define INV_TEMP (1.0f / 0.3f)
// exp(x/T) == 2^(x * log2(e)/T): fold both constants into one mul
// before the native base-2 v_exp_f32.
#define EXP2_SCALE 4.8089834696298783f
#define COLS_PER_BLOCK 2048
#define NCHUNK (NCOL / COLS_PER_BLOCK) // 100

// ---------------------------------------------------------------
// Kernel 1: L2-normalize features; emit fp32 (for exact pos-logit)
// and bf16 (WMMA A operand).
// ---------------------------------------------------------------
__global__ __launch_bounds__(128) void mb_normalize(
    const float* __restrict__ feat,
    float* __restrict__ featN,
    unsigned short* __restrict__ featB_u) {
  __bf16* featB = (__bf16*)featB_u;
  const int b = blockIdx.x;
  const int d = threadIdx.x;           // 128 threads = 4 waves
  float x = feat[b * D_K + d];
  __shared__ float ss[D_K];
  ss[d] = x * x;
  __syncthreads();
  for (int o = 64; o > 0; o >>= 1) {
    if (d < o) ss[d] += ss[d + o];
    __syncthreads();
  }
  float nrm = fmaxf(sqrtf(ss[0]), 1e-12f);
  float y = x / nrm;
  featN[b * D_K + d] = y;
  featB[b * D_K + d] = (__bf16)y;
}

// ---------------------------------------------------------------
// Kernel 2: one-shot fp32 -> bf16 conversion of the memory bank
// (52 MB bf16; stays resident in the 192 MB L2 for the GEMM).
// ---------------------------------------------------------------
__global__ __launch_bounds__(256) void mb_cvt_mem(
    const float4* __restrict__ src,
    unsigned short* __restrict__ dst_u, int n4) {
  int i = blockIdx.x * blockDim.x + threadIdx.x;
  if (i < n4) {
    float4 v = src[i];
    v4bf o = { (__bf16)v.x, (__bf16)v.y, (__bf16)v.z, (__bf16)v.w };
    ((v4bf*)dst_u)[i] = o;
  }
}

// ---------------------------------------------------------------
// Kernel 3: tiled bf16 WMMA GEMM (sims = feat . mem^T / T) fused
// with per-row sum-of-exp.  Block = 256 thr (8 waves); each wave
// owns the block's 16-row tile and 16 disjoint 16-col tiles.
// Software-pipelined: B fragments for tile t+1 stream in while the
// 4 x v_wmma_f32_16x16x32_bf16 + 8 x v_exp of tile t execute, so
// the pre-WMMA s_wait_loadcnt keeps 8 loads in flight instead of
// draining to zero.
// ---------------------------------------------------------------
__global__ __launch_bounds__(256) void mb_gemm_sumexp(
    const unsigned short* __restrict__ featB_u,
    const unsigned short* __restrict__ memB_u,
    float* __restrict__ Spart) {
  const __bf16* featB = (const __bf16*)featB_u;
  const __bf16* memB  = (const __bf16*)memB_u;

  const int lane = threadIdx.x & 31;
  const int wave = threadIdx.x >> 5;
  const int grp  = lane >> 4;          // half-wave select
  const int l15  = lane & 15;
  const int m0   = blockIdx.y * 16;    // row tile base
  const int colBase = blockIdx.x * COLS_PER_BLOCK;

  // ---- A fragments: 16x32 bf16, contiguous K order, same as B ----
  // lanes 0-15 : halves 0-15 -> K 0-15
  // lanes 16-31: halves 0-15 -> K 16-31
  v16bf a[4];
  {
    const __bf16* arow = featB + (size_t)(m0 + l15) * D_K;
    #pragma unroll
    for (int kc = 0; kc < 4; ++kc) {
      union { v16bf v; v8bf h[2]; } u;
      u.h[0] = *(const v8bf*)(arow + kc * 32 + grp * 16);
      u.h[1] = *(const v8bf*)(arow + kc * 32 + grp * 16 + 8);
      a[kc] = u.v;
    }
  }

  float s[8];
  #pragma unroll
  for (int j = 0; j < 8; ++j) s[j] = 0.0f;

  const int TILES = COLS_PER_BLOCK / (16 * 8);          // 16 col tiles/wave
  const size_t BSTRIDE = (size_t)8 * 16 * D_K;          // elements per t-step

  // ---- B fragment pointer: 32x16 bf16; lane = column, 16
  // contiguous K per half-wave -> one aligned 32B load per chunk.
  const __bf16* brow =
      memB + (size_t)(colBase + wave * 16 + l15) * D_K + (size_t)grp * 16;

  v16bf bcur[4], bnxt[4];
  #pragma unroll
  for (int kc = 0; kc < 4; ++kc)
    bcur[kc] = *(const v16bf*)(brow + kc * 32);

  #pragma unroll 2
  for (int t = 0; t < TILES; ++t) {
    const __bf16* bnext_p = brow + BSTRIDE;
    if (t + 1 < TILES) {
      #pragma unroll
      for (int kc = 0; kc < 4; ++kc)
        bnxt[kc] = *(const v16bf*)(bnext_p + kc * 32);
      __builtin_prefetch(bnext_p + BSTRIDE, 0, 1);       // tile t+2 -> L2
    }

    v8f acc = {};
    #pragma unroll
    for (int kc = 0; kc < 4; ++kc)
      acc = __builtin_amdgcn_wmma_f32_16x16x32_bf16(
          /*neg_a=*/false, a[kc], /*neg_b=*/false, bcur[kc],
          /*c_mod=*/(short)0, acc, /*reuse_a=*/false, /*reuse_b=*/false);

    // Epilogue: 1 mul + 1 native exp2 per logit.  Max logit/T ~ 48,
    // exp(48)=7e20, sum over 2e5 << f32 max -> shift-free sum-exp
    // is overflow-safe for this workload.
    #pragma unroll
    for (int j = 0; j < 8; ++j)
      s[j] += __builtin_amdgcn_exp2f(acc[j] * EXP2_SCALE);

    #pragma unroll
    for (int kc = 0; kc < 4; ++kc) bcur[kc] = bnxt[kc];
    brow = bnext_p;
  }

  // Row sums: reduce across the 16 column-lanes of each half-wave.
  #pragma unroll
  for (int j = 0; j < 8; ++j) {
    float v = s[j];
    v += __shfl_xor(v, 1, 32);
    v += __shfl_xor(v, 2, 32);
    v += __shfl_xor(v, 4, 32);
    v += __shfl_xor(v, 8, 32);
    s[j] = v;
  }

  // Combine the 8 waves of the block via LDS, one write per row.
  __shared__ float red[8][16];
  if (l15 == 0) {
    #pragma unroll
    for (int j = 0; j < 8; ++j) red[wave][grp * 8 + j] = s[j];
  }
  __syncthreads();
  if (threadIdx.x < 16) {
    float tsum = 0.0f;
    #pragma unroll
    for (int w = 0; w < 8; ++w) tsum += red[w][threadIdx.x];
    Spart[(size_t)(m0 + threadIdx.x) * NCHUNK + blockIdx.x] = tsum;
  }
}

// ---------------------------------------------------------------
// Kernel 4: finalize.  Per sample: S = sum partials, lse = log S,
// pos = fp32 dot(featN[b], memory[label,0]) / T, weighted mean via
// fixed-order LDS tree (deterministic, no atomics).
// ---------------------------------------------------------------
__global__ __launch_bounds__(1024) void mb_finalize(
    const float* __restrict__ Spart,
    const float* __restrict__ featN,
    const float* __restrict__ memF,
    const int* __restrict__ labels,
    float* __restrict__ out) {
  const int b = threadIdx.x;           // 1024 threads = 1024 samples
  float S = 0.0f;
  #pragma unroll 4
  for (int c = 0; c < NCHUNK; ++c) S += Spart[(size_t)b * NCHUNK + c];
  float lse = logf(S);

  const int lab = labels[b];
  const float* fr = featN + (size_t)b * D_K;
  const float* mr = memF + (size_t)lab * KCAP * D_K;  // bank row 0
  float dot = 0.0f;
  #pragma unroll 8
  for (int d = 0; d < D_K; ++d) dot = fmaf(fr[d], mr[d], dot);

  float w = (lab <= 1) ? 1.3f : 1.0f;  // TAIL_CLASSES = (0, 1)
  float v = w * (lse - dot * INV_TEMP);

  __shared__ float ss[B_N];
  ss[b] = v;
  __syncthreads();
  for (int o = 512; o > 0; o >>= 1) {
    if (b < o) ss[b] += ss[b + o];
    __syncthreads();
  }
  if (b == 0) out[0] = ss[0] / (float)B_N;
}

// ---------------------------------------------------------------
extern "C" void kernel_launch(void* const* d_in, const int* in_sizes, int n_in,
                              void* d_out, int out_size, void* d_ws, size_t ws_size,
                              hipStream_t stream) {
  const float* feat   = (const float*)d_in[0];   // [1024,128] f32
  const int*   labels = (const int*)d_in[1];     // [1024] i32
  const float* memF   = (const float*)d_in[2];   // [50,4096,128] f32

  char* ws = (char*)d_ws;
  size_t off = 0;
  unsigned short* memB  = (unsigned short*)(ws + off); off += (size_t)NCOL * D_K * 2;  // 52,428,800
  float*          featN = (float*)(ws + off);          off += (size_t)B_N * D_K * 4;   // 524,288
  unsigned short* featB = (unsigned short*)(ws + off); off += (size_t)B_N * D_K * 2;   // 262,144
  float*          Spart = (float*)(ws + off);          off += (size_t)B_N * NCHUNK * 4;// 409,600

  mb_normalize<<<B_N, D_K, 0, stream>>>(feat, featN, featB);

  const int n4 = NCOL * D_K / 4;                       // 6,553,600 float4s
  mb_cvt_mem<<<(n4 + 255) / 256, 256, 0, stream>>>((const float4*)memF, memB, n4);

  dim3 grid(NCHUNK, B_N / 16);                          // (100, 64)
  mb_gemm_sumexp<<<grid, 256, 0, stream>>>(featB, memB, Spart);

  mb_finalize<<<1, B_N, 0, stream>>>(Spart, featN, memF, labels, (float*)d_out);
}
